// EdgeMP_69415261438104
// MI455X (gfx1250) — matrix operand order repaired and checked by reading, not compiled, not run on hardware
//
#include <hip/hip_runtime.h>
#include <hip/hip_bf16.h>

typedef __bf16 v16bf __attribute__((ext_vector_type(16)));
typedef float  v8f   __attribute__((ext_vector_type(8)));
typedef unsigned int v4u __attribute__((ext_vector_type(4)));

union Frag16 {            // 32 bytes: one WMMA 16-bit A/B fragment per lane
    v4u   q[2];
    v16bf v;
};

#define F_DIM 128
#define LN_EPS 1e-5f
#define SELU_SCALE 1.0507009873554805f
#define SELU_AS    1.7580993408473766f   // alpha * scale

__device__ __forceinline__ unsigned short f2bf(float f) {
    unsigned int u = __float_as_uint(f);
    u = (u + 0x7FFFu + ((u >> 16) & 1u)) >> 16;
    return (unsigned short)u;
}

// ---------------------------------------------------------------------------
// Pack f32 weights W[K][128] into bf16 B-fragment-major layout:
// fragment (kt,nt): 32 lanes x 16 halves; lane l -> N = nt*16+(l&15),
// K = kt*32 + (l>>4)*8 + j + (j&8)  (j=0..15) per CDNA5 16-bit B layout.
// ---------------------------------------------------------------------------
__global__ void pack_w_kernel(const float* __restrict__ W, int K,
                              unsigned short* __restrict__ out) {
    int t = blockIdx.x * blockDim.x + threadIdx.x;
    int total = (K / 32) * 8 * 32;
    if (t >= total) return;
    int lane = t & 31;
    int f    = t >> 5;
    int nt   = f & 7;
    int kt   = f >> 3;
    int n    = nt * 16 + (lane & 15);
    int kb   = kt * 32 + (lane >> 4) * 8;
    unsigned short* o = out + (size_t)t * 16;
#pragma unroll
    for (int j = 0; j < 16; ++j)
        o[j] = f2bf(W[(size_t)(kb + j + (j & 8)) * F_DIM + n]);
}

// ---------------------------------------------------------------------------
// Count occurrences of col indices (for scatter-mean denominator).
// ---------------------------------------------------------------------------
__global__ void count_kernel(const long long* __restrict__ aidx,
                             float* __restrict__ cnt, long long NA) {
    long long t = (long long)blockIdx.x * blockDim.x + threadIdx.x;
    if (t < NA) atomicAdd(&cnt[aidx[NA + t]], 1.0f);
}

// ---------------------------------------------------------------------------
// Angle MLP: x = concat(a, e[row], e[col]) [384] -> 128 SELU -> 128 -> LN.
// Writes a_new, scatter-adds into sbuf[col].  64 rows / block, 4 waves.
// ---------------------------------------------------------------------------
__global__ void __launch_bounds__(128)
angle_kernel(const float* __restrict__ a, const float* __restrict__ e,
             const long long* __restrict__ aidx,
             const unsigned short* __restrict__ w1p,
             const unsigned short* __restrict__ w2p,
             const float* __restrict__ b1, const float* __restrict__ b2,
             const float* __restrict__ g, const float* __restrict__ beta,
             float* __restrict__ a_new, float* __restrict__ sbuf,
             long long NA) {
    __shared__ __align__(16) unsigned short xbuf[64 * 384];
    __shared__ int rowsh[64];
    __shared__ int colsh[64];

    const int tileBase = blockIdx.x * 64;
    const int tid = threadIdx.x;

    for (int i = tid; i < 64; i += 128) {
        rowsh[i] = (int)aidx[tileBase + i];
        colsh[i] = (int)aidx[NA + tileBase + i];
    }
    __syncthreads();

    // Stage gathered + concatenated input tile as bf16.
    for (int i = tid; i < 64 * 384; i += 128) {
        int rl = i / 384;
        int k  = i - rl * 384;
        float v;
        if (k < 128) {
            v = a[(long long)(tileBase + rl) * F_DIM + k];
        } else {
            long long er = (k < 256) ? (long long)rowsh[rl] : (long long)colsh[rl];
            v = e[er * F_DIM + (k & 127)];
        }
        xbuf[i] = f2bf(v);
    }
    __syncthreads();

    const int wave = tid >> 5, lane = tid & 31;
    const int l15 = lane & 15, hi = lane >> 4;
    unsigned short* xr = xbuf + (size_t)(wave * 16) * 384;
    const Frag16* wp1 = (const Frag16*)w1p;
    const Frag16* wp2 = (const Frag16*)w2p;

    // ---- layer 1: [16x384] @ [384x128] ----
    v8f acc[8];
#pragma unroll
    for (int nt = 0; nt < 8; ++nt) {
        float b = b1[nt * 16 + l15];
#pragma unroll
        for (int r = 0; r < 8; ++r) acc[nt][r] = b;
    }
#pragma unroll
    for (int kt = 0; kt < 12; ++kt) {
        Frag16 af;
        const unsigned short* ap = xr + l15 * 384 + kt * 32 + hi * 8;
        af.q[0] = *(const v4u*)(ap);
        af.q[1] = *(const v4u*)(ap + 16);
#pragma unroll
        for (int nt = 0; nt < 8; ++nt) {
            Frag16 bf = wp1[(kt * 8 + nt) * 32 + lane];
            acc[nt] = __builtin_amdgcn_wmma_f32_16x16x32_bf16(
                false, af.v, false, bf.v, (short)0, acc[nt], false, false);
        }
    }

    // ---- SELU, bf16, back to LDS (reuse region, stride 384) ----
#pragma unroll
    for (int nt = 0; nt < 8; ++nt)
#pragma unroll
        for (int r = 0; r < 8; ++r) {
            float v = acc[nt][r];
            v = v > 0.0f ? SELU_SCALE * v : SELU_AS * (__expf(v) - 1.0f);
            xr[(size_t)(r + hi * 8) * 384 + nt * 16 + l15] = f2bf(v);
        }

    // ---- layer 2: [16x128] @ [128x128] ----
    v8f acc2[8];
#pragma unroll
    for (int nt = 0; nt < 8; ++nt) {
        float b = b2[nt * 16 + l15];
#pragma unroll
        for (int r = 0; r < 8; ++r) acc2[nt][r] = b;
    }
#pragma unroll
    for (int kt = 0; kt < 4; ++kt) {
        Frag16 af;
        const unsigned short* ap = xr + l15 * 384 + kt * 32 + hi * 8;
        af.q[0] = *(const v4u*)(ap);
        af.q[1] = *(const v4u*)(ap + 16);
#pragma unroll
        for (int nt = 0; nt < 8; ++nt) {
            Frag16 bf = wp2[(kt * 8 + nt) * 32 + lane];
            acc2[nt] = __builtin_amdgcn_wmma_f32_16x16x32_bf16(
                false, af.v, false, bf.v, (short)0, acc2[nt], false, false);
        }
    }

    // ---- LayerNorm (rows r + 8*hi; reduce across the 16-lane half) ----
    float s1[8], s2[8];
#pragma unroll
    for (int r = 0; r < 8; ++r) { s1[r] = 0.0f; s2[r] = 0.0f; }
#pragma unroll
    for (int nt = 0; nt < 8; ++nt)
#pragma unroll
        for (int r = 0; r < 8; ++r) {
            float v = acc2[nt][r];
            s1[r] += v;
            s2[r] += v * v;
        }
#pragma unroll
    for (int m = 1; m < 16; m <<= 1)
#pragma unroll
        for (int r = 0; r < 8; ++r) {
            s1[r] += __shfl_xor(s1[r], m, 32);
            s2[r] += __shfl_xor(s2[r], m, 32);
        }
    float mu[8], rstd[8];
#pragma unroll
    for (int r = 0; r < 8; ++r) {
        mu[r] = s1[r] * (1.0f / 128.0f);
        float var = s2[r] * (1.0f / 128.0f) - mu[r] * mu[r];
        rstd[r] = rsqrtf(var + LN_EPS);
    }

    // ---- store a_new + scatter-add into sbuf[col] ----
#pragma unroll
    for (int nt = 0; nt < 8; ++nt) {
        int   col = nt * 16 + l15;
        float gg = g[col], bb = beta[col];
#pragma unroll
        for (int r = 0; r < 8; ++r) {
            int   rl = wave * 16 + r + hi * 8;
            float y  = (acc2[nt][r] - mu[r]) * rstd[r] * gg + bb;
            a_new[(long long)(tileBase + rl) * F_DIM + col] = y;
            atomicAdd(&sbuf[(long long)colsh[rl] * F_DIM + col], y);
        }
    }
}

// ---------------------------------------------------------------------------
// Edge MLP: x = concat(aggr, e) [256] -> 128 SELU -> 128 -> LN.
// aggr read from sbuf (== e_new region); result overwrites same rows.
// ---------------------------------------------------------------------------
__global__ void __launch_bounds__(128)
edge_kernel(const float* __restrict__ e, const float* __restrict__ cnt,
            const unsigned short* __restrict__ w1p,
            const unsigned short* __restrict__ w2p,
            const float* __restrict__ b1, const float* __restrict__ b2,
            const float* __restrict__ g, const float* __restrict__ beta,
            float* __restrict__ e_new /* also sbuf */) {
    __shared__ __align__(16) unsigned short xbuf[64 * 256];
    __shared__ float scl[64];

    const int tileBase = blockIdx.x * 64;
    const int tid = threadIdx.x;

    for (int i = tid; i < 64; i += 128)
        scl[i] = 1.0f / fmaxf(cnt[tileBase + i], 1.0f);
    __syncthreads();

    for (int i = tid; i < 64 * 256; i += 128) {
        int rl = i >> 8;
        int k  = i & 255;
        long long er = tileBase + rl;
        float v = (k < 128) ? e_new[er * F_DIM + k] * scl[rl]
                            : e[er * F_DIM + (k - 128)];
        xbuf[i] = f2bf(v);
    }
    __syncthreads();

    const int wave = tid >> 5, lane = tid & 31;
    const int l15 = lane & 15, hi = lane >> 4;
    unsigned short* xr = xbuf + (size_t)(wave * 16) * 256;
    const Frag16* wp1 = (const Frag16*)w1p;
    const Frag16* wp2 = (const Frag16*)w2p;

    v8f acc[8];
#pragma unroll
    for (int nt = 0; nt < 8; ++nt) {
        float b = b1[nt * 16 + l15];
#pragma unroll
        for (int r = 0; r < 8; ++r) acc[nt][r] = b;
    }
#pragma unroll
    for (int kt = 0; kt < 8; ++kt) {
        Frag16 af;
        const unsigned short* ap = xr + l15 * 256 + kt * 32 + hi * 8;
        af.q[0] = *(const v4u*)(ap);
        af.q[1] = *(const v4u*)(ap + 16);
#pragma unroll
        for (int nt = 0; nt < 8; ++nt) {
            Frag16 bf = wp1[(kt * 8 + nt) * 32 + lane];
            acc[nt] = __builtin_amdgcn_wmma_f32_16x16x32_bf16(
                false, af.v, false, bf.v, (short)0, acc[nt], false, false);
        }
    }

#pragma unroll
    for (int nt = 0; nt < 8; ++nt)
#pragma unroll
        for (int r = 0; r < 8; ++r) {
            float v = acc[nt][r];
            v = v > 0.0f ? SELU_SCALE * v : SELU_AS * (__expf(v) - 1.0f);
            xr[(size_t)(r + hi * 8) * 256 + nt * 16 + l15] = f2bf(v);
        }

    v8f acc2[8];
#pragma unroll
    for (int nt = 0; nt < 8; ++nt) {
        float b = b2[nt * 16 + l15];
#pragma unroll
        for (int r = 0; r < 8; ++r) acc2[nt][r] = b;
    }
#pragma unroll
    for (int kt = 0; kt < 4; ++kt) {
        Frag16 af;
        const unsigned short* ap = xr + l15 * 256 + kt * 32 + hi * 8;
        af.q[0] = *(const v4u*)(ap);
        af.q[1] = *(const v4u*)(ap + 16);
#pragma unroll
        for (int nt = 0; nt < 8; ++nt) {
            Frag16 bf = wp2[(kt * 8 + nt) * 32 + lane];
            acc2[nt] = __builtin_amdgcn_wmma_f32_16x16x32_bf16(
                false, af.v, false, bf.v, (short)0, acc2[nt], false, false);
        }
    }

    float s1[8], s2[8];
#pragma unroll
    for (int r = 0; r < 8; ++r) { s1[r] = 0.0f; s2[r] = 0.0f; }
#pragma unroll
    for (int nt = 0; nt < 8; ++nt)
#pragma unroll
        for (int r = 0; r < 8; ++r) {
            float v = acc2[nt][r];
            s1[r] += v;
            s2[r] += v * v;
        }
#pragma unroll
    for (int m = 1; m < 16; m <<= 1)
#pragma unroll
        for (int r = 0; r < 8; ++r) {
            s1[r] += __shfl_xor(s1[r], m, 32);
            s2[r] += __shfl_xor(s2[r], m, 32);
        }
    float mu[8], rstd[8];
#pragma unroll
    for (int r = 0; r < 8; ++r) {
        mu[r] = s1[r] * (1.0f / 128.0f);
        float var = s2[r] * (1.0f / 128.0f) - mu[r] * mu[r];
        rstd[r] = rsqrtf(var + LN_EPS);
    }
#pragma unroll
    for (int nt = 0; nt < 8; ++nt) {
        int   col = nt * 16 + l15;
        float gg = g[col], bb = beta[col];
#pragma unroll
        for (int r = 0; r < 8; ++r) {
            int rl = wave * 16 + r + hi * 8;
            e_new[(long long)(tileBase + rl) * F_DIM + col] =
                (acc2[nt][r] - mu[r]) * rstd[r] * gg + bb;
        }
    }
}

// ---------------------------------------------------------------------------
extern "C" void kernel_launch(void* const* d_in, const int* in_sizes, int n_in,
                              void* d_out, int out_size, void* d_ws, size_t ws_size,
                              hipStream_t stream) {
    const float*     e     = (const float*)d_in[0];
    const float*     a     = (const float*)d_in[1];
    const long long* aidx  = (const long long*)d_in[2];
    const float*     aw1   = (const float*)d_in[3];
    const float*     ab1   = (const float*)d_in[4];
    const float*     aw2   = (const float*)d_in[5];
    const float*     ab2   = (const float*)d_in[6];
    const float*     ag    = (const float*)d_in[7];
    const float*     abeta = (const float*)d_in[8];
    const float*     ew1   = (const float*)d_in[9];
    const float*     eb1   = (const float*)d_in[10];
    const float*     ew2   = (const float*)d_in[11];
    const float*     eb2   = (const float*)d_in[12];
    const float*     eg    = (const float*)d_in[13];
    const float*     ebeta = (const float*)d_in[14];

    const long long E  = in_sizes[0] / F_DIM;   // 262144
    const long long NA = in_sizes[1] / F_DIM;   // 1048576

    float* out   = (float*)d_out;
    float* e_new = out;                 // doubles as segment-sum buffer
    float* a_new = out + E * F_DIM;

    char* ws = (char*)d_ws;
    float* cnt = (float*)ws;
    unsigned short* w1a = (unsigned short*)(ws + E * 4);
    unsigned short* w2a = w1a + 384 * 128;
    unsigned short* w1e = w2a + 128 * 128;
    unsigned short* w2e = w1e + 256 * 128;

    hipMemsetAsync(e_new, 0, (size_t)E * F_DIM * sizeof(float), stream);
    hipMemsetAsync(cnt, 0, (size_t)E * sizeof(float), stream);

    pack_w_kernel<<<(12 * 8 * 32 + 255) / 256, 256, 0, stream>>>(aw1, 384, w1a);
    pack_w_kernel<<<(4  * 8 * 32 + 255) / 256, 256, 0, stream>>>(aw2, 128, w2a);
    pack_w_kernel<<<(8  * 8 * 32 + 255) / 256, 256, 0, stream>>>(ew1, 256, w1e);
    pack_w_kernel<<<(4  * 8 * 32 + 255) / 256, 256, 0, stream>>>(ew2, 128, w2e);

    count_kernel<<<(int)((NA + 255) / 256), 256, 0, stream>>>(aidx, cnt, NA);

    angle_kernel<<<(int)(NA / 64), 128, 0, stream>>>(
        a, e, aidx, w1a, w2a, ab1, ab2, ag, abeta, a_new, e_new, NA);

    edge_kernel<<<(int)(E / 64), 128, 0, stream>>>(
        e, cnt, w1e, w2e, eb1, eb2, eg, ebeta, e_new);
}
